// DGCNN_21955872817457
// MI455X (gfx1250) — matrix operand order repaired and checked by reading, not compiled, not run on hardware
//
#include <hip/hip_runtime.h>
#include <hip/hip_bf16.h>

// ---------------------------------------------------------------------------
// DGCNN forward for MI455X (gfx1250): bf16 WMMA + async LDS staging.
//   B=2048 N=62 F=512 H=128 K=2 C=2
// Algebra: h2 = A2 * (X * W'^T) + r*cw^T + lin_b   (BN folded into W', cw)
// then 3x (GEMM+bias+relu) + final 992->2 layer.
// ---------------------------------------------------------------------------

typedef __bf16 bf16;
typedef __attribute__((ext_vector_type(16))) __bf16 v16bf;
typedef __attribute__((ext_vector_type(8)))  __bf16 v8bf;
typedef __attribute__((ext_vector_type(8)))  float  v8f;
typedef __attribute__((ext_vector_type(4)))  float  v4f;
typedef __attribute__((ext_vector_type(4)))  int    v4i;

#define WMMA_BF16(a, b, c) \
  __builtin_amdgcn_wmma_f32_16x16x32_bf16(false, (a), false, (b), (short)0, (c), false, false)

__device__ __forceinline__ v16bf frag_cat(v8bf lo, v8bf hi) {
  return __builtin_shufflevector(lo, hi, 0,1,2,3,4,5,6,7,8,9,10,11,12,13,14,15);
}

// 16 bf16 from two 16B chunks (b128 loads)
__device__ __forceinline__ v16bf ld16bf(const bf16* p0, const bf16* p1) {
  v8bf lo = *(const v8bf*)p0;
  v8bf hi = *(const v8bf*)p1;
  return frag_cat(lo, hi);
}

// 16 f32 (two chunks of 8) -> 16 bf16
__device__ __forceinline__ v16bf ld16f32_cvt(const float* p0, const float* p1) {
  v4f a = *(const v4f*)p0;
  v4f b = *(const v4f*)(p0 + 4);
  v4f c = *(const v4f*)p1;
  v4f d = *(const v4f*)(p1 + 4);
  v16bf r;
#pragma unroll
  for (int i = 0; i < 4; ++i) {
    r[i]      = (bf16)a[i];
    r[i + 4]  = (bf16)b[i];
    r[i + 8]  = (bf16)c[i];
    r[i + 12] = (bf16)d[i];
  }
  return r;
}

// ---------------------------------------------------------------------------
// CDNA5 async global->LDS copy (ASYNCcnt tracked), with portable fallback.
// Probe-learned signature: (int4 addrspace(1)* src, int4 addrspace(3)* dst,
//                           imm offset, imm cpol)
// ---------------------------------------------------------------------------
#if __has_builtin(__builtin_amdgcn_global_load_async_to_lds_b128)
#define HAVE_ASYNC_LDS 1
#else
#define HAVE_ASYNC_LDS 0
#endif

typedef __attribute__((address_space(1))) v4i gas_v4i;
typedef __attribute__((address_space(3))) v4i las_v4i;

__device__ __forceinline__ void async_cp16(const void* gsrc, void* ldst) {
#if HAVE_ASYNC_LDS
  __builtin_amdgcn_global_load_async_to_lds_b128((gas_v4i*)gsrc, (las_v4i*)ldst, 0, 0);
#else
  struct alignas(16) B16 { int x[4]; };
  *(B16*)ldst = *(const B16*)gsrc;
#endif
}

__device__ __forceinline__ void wait_async0() {
#if HAVE_ASYNC_LDS
#if __has_builtin(__builtin_amdgcn_s_wait_asynccnt)
  __builtin_amdgcn_s_wait_asynccnt(0);
#else
  asm volatile("s_wait_asynccnt 0x0" ::: "memory");
#endif
#endif
}

// ---------------------------------------------------------------------------
// Kernel 1: adjacency -> A2 = A@A (bf16, padded 64x64), r = rowsum(A2)
// ---------------------------------------------------------------------------
__global__ void k_adj(const float* __restrict__ ew, bf16* __restrict__ A2out,
                      float* __restrict__ rvec) {
  __shared__ float A[64][64];
  __shared__ float dv[64];
  const int t = threadIdx.x;  // 64 threads
#pragma unroll 4
  for (int j = 0; j < 64; ++j) A[t][j] = 0.0f;
  __syncthreads();
  if (t < 62) {
    int base = t * (t + 1) / 2;
    for (int j = 0; j <= t; ++j) {
      float w = ew[base + j];
      A[t][j] = w;
      A[j][t] = w;  // symmetric; diagonal written with same value
    }
  }
  __syncthreads();
  // relu + first sym normalization
  float sm = 0.0f;
  for (int j = 0; j < 64; ++j) {
    float v = fmaxf(A[t][j], 0.0f);
    A[t][j] = v;
    sm += v;
  }
  dv[t] = rsqrtf(sm + 1e-10f);
  __syncthreads();
  float deg = 0.0f;
  for (int j = 0; j < 64; ++j) {
    float v = A[t][j] * dv[t] * dv[j];
    A[t][j] = v;
    deg += v;
  }
  float dis = (deg > 0.0f) ? rsqrtf(deg) : 0.0f;
  dv[t] = dis;
  __syncthreads();
  for (int j = 0; j < 64; ++j) A[t][j] *= dv[t] * dv[j];
  __syncthreads();
  // A2 = A @ A, r = rowsum
  float rs = 0.0f;
  for (int j = 0; j < 64; ++j) {
    float s = 0.0f;
    for (int k = 0; k < 64; ++k) s += A[t][k] * A[k][j];
    A2out[t * 64 + j] = (bf16)s;
    rs += s;
  }
  rvec[t] = rs;
}

// ---------------------------------------------------------------------------
// Kernel 2: BN statistics (sum / sumsq per feature, F=512, over B*N rows)
// ---------------------------------------------------------------------------
__global__ void k_bnstats(const float* __restrict__ x, float* __restrict__ sums,
                          float* __restrict__ sqs) {
  const int R = 2048 * 62;
  int rows_per = (R + gridDim.x - 1) / gridDim.x;
  int r0 = blockIdx.x * rows_per;
  int r1 = min(r0 + rows_per, R);
  int t = threadIdx.x;  // 256 threads -> features t and t+256
  float s0 = 0.f, q0 = 0.f, s1 = 0.f, q1 = 0.f;
  for (int r = r0; r < r1; ++r) {
    const float* p = x + (size_t)r * 512;
    float a = p[t];
    float b = p[t + 256];
    s0 += a; q0 += a * a;
    s1 += b; q1 += b * b;
  }
  atomicAdd(&sums[t], s0);
  atomicAdd(&sqs[t], q0);
  atomicAdd(&sums[t + 256], s1);
  atomicAdd(&sqs[t + 256], q1);
}

// ---------------------------------------------------------------------------
// Kernel 3: fold BN into W' (bf16 [128,512]) and cw[h] = lin_w[h,:] . t
// ---------------------------------------------------------------------------
__global__ void k_bnfinish(const float* __restrict__ sums, const float* __restrict__ sqs,
                           const float* __restrict__ gamma, const float* __restrict__ beta,
                           const float* __restrict__ lin_w,
                           bf16* __restrict__ Wlp, float* __restrict__ cw) {
  __shared__ float ts[512];
  const int f = threadIdx.x;  // 512 threads
  const float invn = 1.0f / (2048.0f * 62.0f);
  float mean = sums[f] * invn;
  float var  = sqs[f] * invn - mean * mean;
  float s    = gamma[f] * rsqrtf(var + 1e-5f);
  float tv   = beta[f] - mean * s;
  ts[f] = tv;
  for (int h = 0; h < 128; ++h)
    Wlp[h * 512 + f] = (bf16)(lin_w[h * 512 + f] * s);
  __syncthreads();
  if (f < 128) {
    float acc = 0.0f;
    for (int k = 0; k < 512; ++k) acc += lin_w[f * 512 + k] * ts[k];
    cw[f] = acc;
  }
}

// ---------------------------------------------------------------------------
// Kernel 4: per-batch fused SGConv:  G = X @ W'^T  (WMMA, fp32->bf16 A),
//           H2 = A2 @ G (WMMA via LDS-transposed G), epilogue + store bf16.
// One block per batch sample: 256 threads = 8 waves, tiles 64x128.
// ---------------------------------------------------------------------------
__launch_bounds__(256)
__global__ void k_stage1(const float* __restrict__ x, const bf16* __restrict__ Wlp,
                         const bf16* __restrict__ A2, const float* __restrict__ rvec,
                         const float* __restrict__ cw, const float* __restrict__ lin_b,
                         bf16* __restrict__ Act0) {
  __shared__ bf16 GT[128 * 72];  // G transposed [h][node], padded stride 72
  const int b    = blockIdx.x;
  const int lane = threadIdx.x & 31;
  const int w    = threadIdx.x >> 5;
  const int wm   = w & 3;    // row tile (16 nodes)
  const int wn   = w >> 2;   // col half (64 h)
  const float* xb = x + (size_t)b * 62 * 512;

  const v8f vz = {0.f, 0.f, 0.f, 0.f, 0.f, 0.f, 0.f, 0.f};
  v8f acc[4] = {vz, vz, vz, vz};

  // ---- stage 1: G[node, h] = X @ W'^T, K = 512
  const int rr = min(wm * 16 + (lane & 15), 61);  // clamp (pad rows harmless: A2 cols 62/63 = 0)
  for (int k0 = 0; k0 < 512; k0 += 32) {
    const float* px = xb + (size_t)rr * 512 + k0 + ((lane & 16) ? 8 : 0);
    v16bf a = ld16f32_cvt(px, px + 16);
#pragma unroll
    for (int ct = 0; ct < 4; ++ct) {
      int hn = wn * 64 + ct * 16 + (lane & 15);
      const bf16* pw = Wlp + (size_t)hn * 512 + k0 + ((lane & 16) ? 8 : 0);
      v16bf bb = ld16bf(pw, pw + 16);
      acc[ct] = WMMA_BF16(a, bb, acc[ct]);
    }
  }
  // write G transposed to LDS (per lane: 8 consecutive nodes, one h column)
  const int nodebase = wm * 16 + ((lane & 16) ? 8 : 0);
#pragma unroll
  for (int ct = 0; ct < 4; ++ct) {
    int hn = wn * 64 + ct * 16 + (lane & 15);
    v8bf pk;
#pragma unroll
    for (int j = 0; j < 8; ++j) pk[j] = (bf16)acc[ct][j];
    *(v8bf*)&GT[hn * 72 + nodebase] = pk;
  }
  __syncthreads();

  // ---- stage 2: H2 = A2 @ G, K = 64
  v8f acc2[4] = {vz, vz, vz, vz};
#pragma unroll
  for (int k0 = 0; k0 < 64; k0 += 32) {
    const bf16* pa = A2 + (size_t)(wm * 16 + (lane & 15)) * 64 + k0 + ((lane & 16) ? 8 : 0);
    v16bf a = ld16bf(pa, pa + 16);
#pragma unroll
    for (int ct = 0; ct < 4; ++ct) {
      int hn = wn * 64 + ct * 16 + (lane & 15);
      const bf16* pg = &GT[hn * 72 + k0 + ((lane & 16) ? 8 : 0)];
      v16bf bb = ld16bf(pg, pg + 16);
      acc2[ct] = WMMA_BF16(a, bb, acc2[ct]);
    }
  }

  // ---- epilogue: + r[node]*cw[h] + lin_b[h]; store bf16 [b, node*128+h]
  const int mrow = wm * 16 + ((lane & 16) ? 8 : 0);
#pragma unroll
  for (int ct = 0; ct < 4; ++ct) {
    int hn = wn * 64 + ct * 16 + (lane & 15);
    float cwv = cw[hn];
    float lbv = lin_b[hn];
#pragma unroll
    for (int j = 0; j < 8; ++j) {
      int node = mrow + j;
      if (node < 62) {
        float v = acc2[ct][j] + rvec[node] * cwv + lbv;
        Act0[(size_t)b * 7936 + node * 128 + hn] = (bf16)v;
      }
    }
  }
}

// ---------------------------------------------------------------------------
// Kernel 5: WMMA GEMM  Out = relu(Act @ W^T + bias), bf16 out, with
// double-buffered async global->LDS staging of both operand tiles.
// A: bf16 [M,K] row-major. W: f32 [N,K] row-major (bf16-converted on read).
// Block tile 256x128: 8 waves, each wave owns 32 rows x 128 cols
// (2 row-tiles x 8 col-tiles = 16 WMMA accums = 128 VGPRs/wave).
// AI = 2*256*128*32 / 32KB = 64 flops/byte of LDS-staged traffic.
// LDS: wbuf padded to 36 f32/row, abuf padded to 40 bf16/row -> the b128
// fragment reads are bank-conflict-free (stride = 4 mod 64 dwords).
// Requires: M % 256 == 0, K % 32 == 0. N masked.
// ---------------------------------------------------------------------------
__launch_bounds__(256)
__global__ void fc_gemm(const bf16* __restrict__ Aact, const float* __restrict__ W,
                        const float* __restrict__ bias, bf16* __restrict__ Out,
                        int M, int N, int K, int do_relu) {
  __shared__ float wbuf[2][128 * 36];  // 2 x 18.0 KB
  __shared__ bf16  abuf[2][256 * 40];  // 2 x 20.0 KB
  const int tid  = threadIdx.x;
  const int lane = tid & 31;
  const int w    = tid >> 5;           // wave 0..7 -> 32-row strip
  const int m0b  = blockIdx.y * 256;
  const int n0b  = blockIdx.x * 128;

  // Stage one 128x32 W tile (fp32) + one 256x32 A tile (bf16) into LDS.
  // W: 1024 x 16B chunks -> 4/thread.  A: 1024 x 16B chunks -> 4/thread.
  auto copy_tile = [&](int buf, int k0) {
#pragma unroll
    for (int i = 0; i < 4; ++i) {
      int c   = tid * 4 + i;
      int row = c >> 3;
      int kc  = (c & 7) * 4;  // float index within 32-wide k slab
      int gr  = min(n0b + row, N - 1);
      async_cp16(W + (size_t)gr * K + k0 + kc, &wbuf[buf][row * 36 + kc]);
    }
#pragma unroll
    for (int i = 0; i < 4; ++i) {
      int c   = tid * 4 + i;
      int row = c >> 2;       // 4 chunks per 32-bf16 row
      int kc  = (c & 3) * 8;  // bf16 index within 32-wide k slab
      async_cp16(Aact + (size_t)(m0b + row) * K + k0 + kc, &abuf[buf][row * 40 + kc]);
    }
  };

  const v8f vz = {0.f, 0.f, 0.f, 0.f, 0.f, 0.f, 0.f, 0.f};
  v8f acc[2][8] = {{vz, vz, vz, vz, vz, vz, vz, vz},
                   {vz, vz, vz, vz, vz, vz, vz, vz}};

  const int nk = K / 32;
  copy_tile(0, 0);
  wait_async0();
  __syncthreads();

  for (int kt = 0; kt < nk; ++kt) {
    const int cur = kt & 1;
    if (kt + 1 < nk) copy_tile(1 - cur, (kt + 1) * 32);  // prefetch next tile

    const bf16* a0p = &abuf[cur][(w * 32 + (lane & 15)) * 40 + ((lane & 16) ? 8 : 0)];
    const bf16* a1p = a0p + 16 * 40;
    v16bf a0 = ld16bf(a0p, a0p + 16);
    v16bf a1 = ld16bf(a1p, a1p + 16);
#pragma unroll
    for (int ct = 0; ct < 8; ++ct) {
      const float* pw = &wbuf[cur][(ct * 16 + (lane & 15)) * 36 + ((lane & 16) ? 8 : 0)];
      v16bf bb = ld16f32_cvt(pw, pw + 16);
      acc[0][ct] = WMMA_BF16(a0, bb, acc[0][ct]);
      acc[1][ct] = WMMA_BF16(a1, bb, acc[1][ct]);
    }

    if (kt + 1 < nk) wait_async0();  // next tile landed while we computed
    __syncthreads();                 // all reads of tile kt done; tile kt+1 visible
  }

#pragma unroll
  for (int rt = 0; rt < 2; ++rt) {
    int mrow = m0b + w * 32 + rt * 16 + ((lane & 16) ? 8 : 0);
#pragma unroll
    for (int ct = 0; ct < 8; ++ct) {
      int cn = n0b + ct * 16 + (lane & 15);
      if (cn < N) {
        float bv = bias[cn];
#pragma unroll
        for (int j = 0; j < 8; ++j) {
          float v = acc[rt][ct][j] + bv;
          if (do_relu) v = fmaxf(v, 0.0f);
          Out[(size_t)(mrow + j) * N + cn] = (bf16)v;
        }
      }
    }
  }
}

// ---------------------------------------------------------------------------
// Kernel 6: final 992 -> 2 layer, fp32 out. One thread per batch row.
// ---------------------------------------------------------------------------
__global__ void k_fc3(const bf16* __restrict__ Act3, const float* __restrict__ W,
                      const float* __restrict__ bias, float* __restrict__ out) {
  int b = blockIdx.x * blockDim.x + threadIdx.x;
  if (b >= 2048) return;
  const bf16* a = Act3 + (size_t)b * 992;
  float a0 = 0.f, a1 = 0.f;
  for (int k = 0; k < 992; ++k) {
    float v = (float)a[k];
    a0 += v * W[k];
    a1 += v * W[992 + k];
  }
  out[b * 2 + 0] = a0 + bias[0];
  out[b * 2 + 1] = a1 + bias[1];
}

// ---------------------------------------------------------------------------
extern "C" void kernel_launch(void* const* d_in, const int* in_sizes, int n_in,
                              void* d_out, int out_size, void* d_ws, size_t ws_size,
                              hipStream_t stream) {
  const float* x     = (const float*)d_in[0];
  const float* ew    = (const float*)d_in[1];
  const float* gamma = (const float*)d_in[2];
  const float* beta  = (const float*)d_in[3];
  const float* lin_w = (const float*)d_in[4];
  const float* lin_b = (const float*)d_in[5];
  const float* fc0w  = (const float*)d_in[6];
  const float* fc0b  = (const float*)d_in[7];
  const float* fc1w  = (const float*)d_in[8];
  const float* fc1b  = (const float*)d_in[9];
  const float* fc2w  = (const float*)d_in[10];
  const float* fc2b  = (const float*)d_in[11];
  const float* fc3w  = (const float*)d_in[12];
  const float* fc3b  = (const float*)d_in[13];

  char* ws = (char*)d_ws;
  float* bn_sum = (float*)(ws + 0);        // 512 f32
  float* bn_sq  = (float*)(ws + 2048);     // 512 f32
  float* cw     = (float*)(ws + 4096);     // 128 f32
  float* rvec   = (float*)(ws + 4608);     // 64 f32
  bf16*  A2     = (bf16*)(ws + 5120);      // 64*64 bf16 = 8192 B
  bf16*  Wlp    = (bf16*)(ws + 13312);     // 128*512 bf16 = 131072 B
  bf16*  Act0   = (bf16*)(ws + 144640);    // 2048*7936 bf16 = 32,505,856 B
  bf16*  Act1   = (bf16*)(ws + 32650496);  // 2048*3968 bf16 = 16,252,928 B
  bf16*  Act2   = (bf16*)(ws + 48903424);  // 2048*1984 bf16 =  8,126,464 B
  bf16*  Act3   = (bf16*)(ws + 57029888);  // 2048*992  bf16 =  4,063,232 B

  // zero the BN accumulators (capture-safe)
  (void)hipMemsetAsync(d_ws, 0, 4096, stream);

  k_adj<<<1, 64, 0, stream>>>(ew, A2, rvec);
  k_bnstats<<<256, 256, 0, stream>>>(x, bn_sum, bn_sq);
  k_bnfinish<<<1, 512, 0, stream>>>(bn_sum, bn_sq, gamma, beta, lin_w, Wlp, cw);
  k_stage1<<<2048, 256, 0, stream>>>(x, Wlp, A2, rvec, cw, lin_b, Act0);

  fc_gemm<<<dim3(31, 8), 256, 0, stream>>>(Act0, fc0w, fc0b, Act1, 2048, 3968, 7936, 1);
  fc_gemm<<<dim3(16, 8), 256, 0, stream>>>(Act1, fc1w, fc1b, Act2, 2048, 1984, 3968, 1);
  fc_gemm<<<dim3(8, 8),  256, 0, stream>>>(Act2, fc2w, fc2b, Act3, 2048, 992, 1984, 1);

  k_fc3<<<8, 256, 0, stream>>>(Act3, fc3w, fc3b, (float*)d_out);
}
